// CUPY_SNN_PLIF_11682311045231
// MI455X (gfx1250) — matrix operand order repaired and checked by reading, not compiled, not run on hardware
//
#include <hip/hip_runtime.h>
#include <math.h>

// ---------------------------------------------------------------------------
// SNN PLIF pipeline for MI455X (gfx1250, wave32).
//
// Roofline: x = 138MB fp32, y would be 277MB. We never spill y to HBM:
// phase A recomputes y per block for BN batch stats, phase C recomputes it
// for the PLIF scan. HBM traffic ~= 2 x 138MB ~= 12us @ 23.3 TB/s, vs ~692MB
// if y round-tripped. Extra compute (~4.6 GMAC) is negligible vs WMMA peak.
//
// encode_C (44x22 @ 22x384 per sample) uses v_wmma_f32_16x16x32_f16 with
// K=22 zero-padded to 32: ONE wmma per 16x16 D tile. Operands are staged in
// LDS pre-padded & pre-transposed so every fragment is built from aligned,
// UNCONDITIONAL ds_load_b128s.
//
// Depthwise conv (K=12) uses a register rolling window: 1 ds_load per output
// point instead of 12 (consecutive t share 11 of 12 taps).
// ---------------------------------------------------------------------------

#define NB     4096
#define CIN    22
#define TT     384
#define CCH    44
#define KW     12
#define PADW   6
#define TOUT   385
#define ONUM   4
#define NSTAT  (4096.0f * 385.0f)
#define NTHREADS 256
#define KPAD   24          // xT k-stride (22 real + 2 zero, 16B-alignable)
#define YSTR   390         // yenc t-stride: 195 dwords, gcd(195,64)=1 -> no bank conflicts
#define CHUNK  77          // 5 chunks x 77 = 385 timesteps (stats phase)

typedef __attribute__((ext_vector_type(16))) _Float16 v16h;
typedef __attribute__((ext_vector_type(8)))  _Float16 v8h;
typedef __attribute__((ext_vector_type(8)))  float    v8f;

struct LdsLayout {
    _Float16 xT[TT * KPAD];    // 18432 B : x_b transposed [t][k], k padded to 24
    _Float16 yenc[48 * YSTR];  // 37440 B : encode_C output (rows 44..47 = pad)
    _Float16 wA[48 * 32];      //  3072 B : wC zero-padded to 48x32 (A operand)
    float    wT[CCH * KW];     //  2112 B : depthwise kernels
    float    stat_sum[CCH];    //   176 B
    float    stat_sq[CCH];     //   176 B
    float    rate[CCH];        //   176 B
};                             // total ~61.6 KB per workgroup

// Stage operands into LDS (padded/transposed), then encode_C with WMMA.
// One workgroup = one sample b; 8 waves cover 24 N-tiles x 3 M-tiles.
__device__ __forceinline__ void load_and_encode(const float* __restrict__ x,
                                                const float* __restrict__ wC,
                                                const float* __restrict__ wT,
                                                LdsLayout& L, int b) {
    const int tid = threadIdx.x;
    const float* xb = x + (size_t)b * (CIN * TT);

    // gfx1250 speculative prefetch of this block's x slice (global_prefetch_b8)
    __builtin_prefetch(xb + tid * 33, 0, 0);

    // x transposed: coalesced global read, scattered LDS write (k contiguous)
    for (int i = tid; i < CIN * TT; i += NTHREADS) {
        const int c = i / TT;
        const int t = i - c * TT;
        L.xT[t * KPAD + c] = (_Float16)xb[i];
    }
    for (int i = tid; i < TT * 2; i += NTHREADS)           // zero k=22,23
        L.xT[(i >> 1) * KPAD + CIN + (i & 1)] = (_Float16)0.0f;

    // wC zero-padded to 48x32 (unconditional load via clamped address)
    for (int i = tid; i < 48 * 32; i += NTHREADS) {
        const int row = i >> 5;
        const int k   = i & 31;
        const bool ok = (row < CCH) && (k < CIN);
        const float wv = wC[ok ? row * CIN + k : 0];
        L.wA[i] = ok ? (_Float16)wv : (_Float16)0.0f;
    }
    for (int i = tid; i < CCH * KW; i += NTHREADS) L.wT[i] = wT[i];
    __syncthreads();

    const int wave  = tid >> 5;
    const int lane  = tid & 31;
    const int lhalf = lane & 15;
    const int hi    = lane >> 4;     // lanes 16..31: "high" K halves

    for (int tn = wave; tn < 24; tn += 8) {
        const int col = tn * 16 + lhalf;

        // B fragment (32x16 KxN): lane holds column `col`, K = 16*hi + i.
        // xT row is 48B: pb[0]=k0..7, pb[1]=k8..15, pb[2]=k16..23.
        const v8h* pb = (const v8h*)&L.xT[col * KPAD];
        v8h b0 = pb[hi ? 2 : 0];
        v8h b1 = pb[1];
        const v8h vz = {};
        b1 = hi ? vz : b1;           // K=24..31 are zero (beyond pad)
        const v16h bf = __builtin_shufflevector(b0, b1,
            0, 1, 2, 3, 4, 5, 6, 7, 8, 9, 10, 11, 12, 13, 14, 15);

        #pragma unroll
        for (int tm = 0; tm < 3; ++tm) {
            const int row = tm * 16 + lhalf;
            // A fragment (16x32 MxK): K = 16*hi + i (contiguous, matches B).
            const v8h* pa = (const v8h*)&L.wA[row * 32 + 16 * hi];
            const v16h af = __builtin_shufflevector(pa[0], pa[1],
                0, 1, 2, 3, 4, 5, 6, 7, 8, 9, 10, 11, 12, 13, 14, 15);

            v8f acc = {};
            acc = __builtin_amdgcn_wmma_f32_16x16x32_f16(
                /*neg_a=*/false, af, /*neg_b=*/false, bf,
                /*c_mod=*/(short)0, acc, /*reuse_a=*/false, /*reuse_b=*/false);

            // D layout: VGPR r, lane n -> (M = r + 8*(n>=16), N = n%16)
            #pragma unroll
            for (int r = 0; r < 8; ++r)
                L.yenc[(tm * 16 + r + 8 * hi) * YSTR + col] = (_Float16)acc[r];
        }
    }
    __syncthreads();
}

// Clamped encode-output read (boundary -> 0, compiles to v_cndmask).
__device__ __forceinline__ float yget(const LdsLayout& L, int c, int ti) {
    return (ti >= 0 && ti < TT) ? (float)L.yenc[c * YSTR + ti] : 0.0f;
}

__global__ void snn_zero_kernel(float* ws) {
    if (threadIdx.x < 128) ws[threadIdx.x] = 0.0f;   // gsum[64] + gsq[64]
}

// Phase A: recompute y per block, accumulate per-channel sum / sum^2.
__global__ __launch_bounds__(NTHREADS)
void snn_stats_kernel(const float* __restrict__ x, const float* __restrict__ wC,
                      const float* __restrict__ wT,
                      float* __restrict__ gsum, float* __restrict__ gsq) {
    __shared__ LdsLayout L;
    const int b = blockIdx.x;
    const int tid = threadIdx.x;
    if (tid < CCH) { L.stat_sum[tid] = 0.0f; L.stat_sq[tid] = 0.0f; }
    load_and_encode(x, wC, wT, L, b);   // barrier inside covers the zero-init

    // 220 threads = 44 channels x 5 contiguous 77-step chunks, so the conv
    // rolling window applies (1 LDS load per output point, not 12).
    if (tid < CCH * 5) {
        const int c  = tid % CCH;
        const int t0 = (tid / CCH) * CHUNK;
        float wreg[KW];
        #pragma unroll
        for (int k = 0; k < KW; ++k) wreg[k] = L.wT[c * KW + k];

        float win[KW - 1];                      // in[t-6 .. t+4]
        #pragma unroll
        for (int k = 0; k < KW - 1; ++k) win[k] = yget(L, c, t0 - PADW + k);

        float ps = 0.0f, pq = 0.0f;
        for (int t = t0; t < t0 + CHUNK; ++t) {
            const float nv = yget(L, c, t + (KW - 1) - PADW);   // in[t+5]
            float y = wreg[KW - 1] * nv;
            #pragma unroll
            for (int k = 0; k < KW - 1; ++k) y += wreg[k] * win[k];
            #pragma unroll
            for (int k = 0; k < KW - 2; ++k) win[k] = win[k + 1];
            win[KW - 2] = nv;
            ps += y;
            pq += y * y;
        }
        atomicAdd(&L.stat_sum[c], ps);   // ds_add_f32
        atomicAdd(&L.stat_sq[c],  pq);
    }
    __syncthreads();
    if (tid < CCH) {
        atomicAdd(&gsum[tid], L.stat_sum[tid]);   // global_atomic_add_f32
        atomicAdd(&gsq[tid],  L.stat_sq[tid]);
    }
}

// Fold BN into per-channel scale/shift.
__global__ void snn_finalize_kernel(const float* __restrict__ gamma,
                                    const float* __restrict__ beta,
                                    const float* __restrict__ gsum,
                                    const float* __restrict__ gsq,
                                    float* __restrict__ scale,
                                    float* __restrict__ shift) {
    const int c = threadIdx.x;
    if (c < CCH) {
        const float mu   = gsum[c] / NSTAT;
        const float var  = gsq[c] / NSTAT - mu * mu;
        const float rstd = rsqrtf(var + 1e-5f);
        const float sc   = gamma[c] * rstd;
        scale[c] = sc;
        shift[c] = beta[c] - mu * sc;
    }
}

// Phase C: recompute y, BN affine, sequential PLIF scan, classifier.
__global__ __launch_bounds__(NTHREADS)
void snn_scan_kernel(const float* __restrict__ x, const float* __restrict__ wC,
                     const float* __restrict__ wT,
                     const float* __restrict__ scale,
                     const float* __restrict__ shift,
                     const float* __restrict__ w_plif,
                     const float* __restrict__ Wcls,
                     const float* __restrict__ bcls,
                     float* __restrict__ out) {
    __shared__ LdsLayout L;
    const int b = blockIdx.x;
    const int tid = threadIdx.x;
    load_and_encode(x, wC, wT, L, b);

    const float decay = 1.0f / (1.0f + __expf(-w_plif[0]));

    if (tid < CCH) {
        const int c = tid;
        const float sc = scale[c];
        const float sh = shift[c];
        float wreg[KW];
        #pragma unroll
        for (int k = 0; k < KW; ++k) wreg[k] = L.wT[c * KW + k];

        float win[KW - 1];                      // in[t-6 .. t+4]
        #pragma unroll
        for (int k = 0; k < KW - 1; ++k) win[k] = yget(L, c, -PADW + k);

        float v = 0.0f, acc = 0.0f;
        for (int t = 0; t < TOUT; ++t) {
            const float nv = yget(L, c, t + (KW - 1) - PADW);   // in[t+5]
            float y = wreg[KW - 1] * nv;
            #pragma unroll
            for (int k = 0; k < KW - 1; ++k) y += wreg[k] * win[k];
            #pragma unroll
            for (int k = 0; k < KW - 2; ++k) win[k] = win[k + 1];
            win[KW - 2] = nv;

            y = y * sc + sh;                            // BN affine
            v += (y - v) * decay;                       // PLIF integrate
            const float s = (v >= 1.0f) ? 1.0f : 0.0f;  // heaviside(v - V_TH)
            acc += s;
            v *= (1.0f - s);                            // hard reset
        }
        L.rate[c] = acc * (1.0f / (float)TOUT);
    }
    __syncthreads();

    if (tid < ONUM) {
        float o = bcls[tid];
        #pragma unroll 4
        for (int c = 0; c < CCH; ++c) o += L.rate[c] * Wcls[tid * CCH + c];
        out[b * ONUM + tid] = o;
    }
}

extern "C" void kernel_launch(void* const* d_in, const int* in_sizes, int n_in,
                              void* d_out, int out_size, void* d_ws, size_t ws_size,
                              hipStream_t stream) {
    const float* x       = (const float*)d_in[0];   // [4096,22,384]
    const float* wC      = (const float*)d_in[1];   // [44,22]
    const float* wT      = (const float*)d_in[2];   // [44,12]
    const float* gamma   = (const float*)d_in[3];   // [44]
    const float* beta_bn = (const float*)d_in[4];   // [44]
    const float* w_plif  = (const float*)d_in[5];   // scalar
    const float* Wcls    = (const float*)d_in[6];   // [4,44]
    const float* bcls    = (const float*)d_in[7];   // [4]

    float* ws    = (float*)d_ws;
    float* gsum  = ws;        // [64] (44 used)
    float* gsq   = ws + 64;   // [64]
    float* scale = ws + 128;  // [64]
    float* shift = ws + 192;  // [64]

    snn_zero_kernel<<<1, 128, 0, stream>>>(ws);
    snn_stats_kernel<<<NB, NTHREADS, 0, stream>>>(x, wC, wT, gsum, gsq);
    snn_finalize_kernel<<<1, 64, 0, stream>>>(gamma, beta_bn, gsum, gsq, scale, shift);
    snn_scan_kernel<<<NB, NTHREADS, 0, stream>>>(x, wC, wT, scale, shift,
                                                 w_plif, Wcls, bcls, (float*)d_out);
}
